// TransformerBlock_16320875725478
// MI455X (gfx1250) — matrix operand order repaired and checked by reading, not compiled, not run on hardware
//
#include <hip/hip_runtime.h>
#include <hip/hip_bf16.h>
#include <math.h>

typedef __attribute__((ext_vector_type(16))) _Float16 v16h;
typedef __attribute__((ext_vector_type(8)))  _Float16 v8h;
typedef __attribute__((ext_vector_type(8)))  float    v8f;

#define DEVI __device__ __forceinline__

static constexpr int B_ = 2, C_ = 256, N_ = 4096, HEADS_ = 8, HD_ = 32, HID_ = 1024, GROUPS_ = 32;

DEVI v16h combine16(v8h lo, v8h hi) {
  v16h r;
#pragma unroll
  for (int i = 0; i < 8; ++i) { r[i] = lo[i]; r[i + 8] = hi[i]; }
  return r;
}

// A fragment (16x32 f16, MxK). Source row-major [m][k], ld = row stride in elements.
// Layout: lanes 0-15 rows M=lane hold K=0..15; lanes 16-31 hold K=16..31.
DEVI v16h load_a_frag(const _Float16* __restrict__ p, int ld) {
  const int lane = threadIdx.x & 31;
  const int m = lane & 15, hv = lane >> 4;
  const _Float16* r = p + (size_t)m * ld;
  v8h lo = *(const v8h*)(r + 16 * hv);
  v8h hi = *(const v8h*)(r + 16 * hv + 8);
  return combine16(lo, hi);
}

// B fragment (32x16 f16, KxN). Source stored [n][k] row-major (k contiguous), ld = row stride.
// Layout: lanes 0-15 (col=lane) hold K=0..15; lanes 16-31 hold K=16..31.
DEVI v16h load_b_frag(const _Float16* __restrict__ p, int ld) {
  const int lane = threadIdx.x & 31;
  const int n = lane & 15, hv = lane >> 4;
  const _Float16* r = p + (size_t)n * ld + 16 * hv;
  v8h lo = *(const v8h*)(r);
  v8h hi = *(const v8h*)(r + 8);
  return combine16(lo, hi);
}

DEVI v8f wmma_f16f32(v16h a, v16h b, v8f c) {
  return __builtin_amdgcn_wmma_f32_16x16x32_f16(false, a, false, b, (short)0, c, false, false);
}

DEVI float half_reduce_max(float v) {
#pragma unroll
  for (int m = 1; m < 16; m <<= 1) v = fmaxf(v, __shfl_xor(v, m, 32));
  return v;
}
DEVI float half_reduce_sum(float v) {
#pragma unroll
  for (int m = 1; m < 16; m <<= 1) v += __shfl_xor(v, m, 32);
  return v;
}

// ---------------- weight convert ----------------
__global__ void cvt_f16_kernel(const float* __restrict__ s, _Float16* __restrict__ d, int n) {
  int i = blockIdx.x * 256 + threadIdx.x;
  if (i < n) d[i] = (_Float16)s[i];
}

// ---------------- GroupNorm: (B,C,N) f32 -> token-major (B,N,C) f16 ----------------
__global__ void groupnorm_kernel(const float* __restrict__ x, const float* __restrict__ w,
                                 const float* __restrict__ bias, _Float16* __restrict__ out) {
  const int b = blockIdx.x >> 5, g = blockIdx.x & 31;
  const int tid = threadIdx.x;
  const int CH = C_ / GROUPS_;          // 8 channels per group
  const int M = CH * N_;                // 32768 elements per group (contiguous)
  const float* base = x + ((size_t)b * C_ + g * CH) * N_;
  float s = 0.f, ss = 0.f;
  for (int i = tid; i < M; i += 256) { float v = base[i]; s += v; ss += v * v; }
  __shared__ float sb[256], sb2[256];
  sb[tid] = s; sb2[tid] = ss; __syncthreads();
  for (int o = 128; o > 0; o >>= 1) {
    if (tid < o) { sb[tid] += sb[tid + o]; sb2[tid] += sb2[tid + o]; }
    __syncthreads();
  }
  const float mu  = sb[0] / (float)M;
  const float var = sb2[0] / (float)M - mu * mu;
  const float inv = rsqrtf(var + 1e-5f);
  for (int j = tid; j < M; j += 256) {
    int c = j & (CH - 1), n = j >> 3;
    int ch = g * CH + c;
    float v = base[(size_t)c * N_ + n];
    out[((size_t)b * N_ + n) * C_ + ch] = (_Float16)(((v - mu) * inv) * w[ch] + bias[ch]);
  }
}

// ---------------- wave-level 32x32 GEMM tile, W(MxK) row-major, Act token-major (N,K) ----------------
struct Tile22 { v8f c00, c01, c10, c11; };

DEVI void gemm_inner(const _Float16* __restrict__ W, const _Float16* __restrict__ A,
                     int K, int row0, int col0, Tile22& t) {
  for (int k0 = 0; k0 < K; k0 += 32) {
    v16h a0 = load_a_frag(W + (size_t)row0 * K + k0, K);
    v16h a1 = load_a_frag(W + (size_t)(row0 + 16) * K + k0, K);
    v16h b0 = load_b_frag(A + (size_t)col0 * K + k0, K);
    v16h b1 = load_b_frag(A + (size_t)(col0 + 16) * K + k0, K);
    t.c00 = wmma_f16f32(a0, b0, t.c00);
    t.c01 = wmma_f16f32(a0, b1, t.c01);
    t.c10 = wmma_f16f32(a1, b0, t.c10);
    t.c11 = wmma_f16f32(a1, b1, t.c11);
  }
}

// ---------------- QKV GEMM: (768x256) x (256x4096) per batch ----------------
// Rows 0-255 -> Q (b,h,n,d); 256-511 -> K (b,h,n,d); 512-767 -> V transposed (b,h,d,n).
__global__ void qkv_gemm_kernel(const _Float16* __restrict__ wq, const _Float16* __restrict__ xh,
                                _Float16* __restrict__ q2, _Float16* __restrict__ k2,
                                _Float16* __restrict__ v2t) {
  const int b = blockIdx.z;
  const int wave = threadIdx.x >> 5, lane = threadIdx.x & 31;
  const int hf = lane >> 4, ln = lane & 15;
  const int row0 = blockIdx.x * 128 + (wave >> 1) * 32;
  const int col0 = blockIdx.y * 64 + (wave & 1) * 32;
  Tile22 t = {};
  gemm_inner(wq, xh + (size_t)b * N_ * C_, C_, row0, col0, t);
  const v8f accs[2][2] = {{t.c00, t.c01}, {t.c10, t.c11}};
#pragma unroll
  for (int i = 0; i < 2; ++i) {
    const int ob = row0 + 16 * i;
    const int s = ob >> 8, hh = (ob >> 5) & 7, d0 = ob & 31;
#pragma unroll
    for (int j = 0; j < 2; ++j) {
      const int n = col0 + 16 * j + ln;
      v8f acc = accs[i][j];
#pragma unroll
      for (int r = 0; r < 8; ++r) {
        const int d = d0 + r + 8 * hf;                 // C/D layout: row = r + 8*half
        const _Float16 val = (_Float16)acc[r];
        if (s == 0)      q2[((size_t)(b * HEADS_ + hh) * N_ + n) * HD_ + d] = val;
        else if (s == 1) k2[((size_t)(b * HEADS_ + hh) * N_ + n) * HD_ + d] = val;
        else             v2t[((size_t)(b * HEADS_ + hh) * HD_ + d) * N_ + n] = val;
      }
    }
  }
}

// ---------------- Flash attention: one wave = 16 query rows, stream 32 keys/iter ----------------
__global__ void flash_attn_kernel(const _Float16* __restrict__ q2, const _Float16* __restrict__ k2,
                                  const _Float16* __restrict__ v2t, const float* __restrict__ phys_bias,
                                  _Float16* __restrict__ attn_out) {
  const int b = blockIdx.z, h = blockIdx.y;
  const int wave = threadIdx.x >> 5, lane = threadIdx.x & 31;
  const int hf = lane >> 4, ln = lane & 15;
  const int n0 = blockIdx.x * 128 + wave * 16;
  const float scale = 0.17677669529663687f;            // 1/sqrt(32)
  const float bias = phys_bias[h];
  const size_t bh = (size_t)(b * HEADS_ + h);

  const v16h aq = load_a_frag(q2 + (bh * N_ + n0) * HD_, HD_);   // Q^T tile: 16 x d(=K=32)
  v8f o0 = {}, o1 = {};
  float Mrun[8], Lrun[8];
#pragma unroll
  for (int r = 0; r < 8; ++r) { Mrun[r] = -3.0e38f; Lrun[r] = 0.f; }

  __shared__ __align__(16) _Float16 plds[8][16 * 40];  // per-wave 16x32 P tile, row stride 40
  _Float16* myp = plds[wave];
  const v8f zero = {};

  for (int m0 = 0; m0 < N_; m0 += 32) {
    const _Float16* kb = k2 + (bh * N_ + m0) * HD_;
    v16h bk0 = load_b_frag(kb, HD_);
    v16h bk1 = load_b_frag(kb + 16 * HD_, HD_);
    v8f s0 = wmma_f16f32(aq, bk0, zero);
    v8f s1 = wmma_f16f32(aq, bk1, zero);

    float p0[8], p1[8];
#pragma unroll
    for (int r = 0; r < 8; ++r) {
      s0[r] = s0[r] * scale + bias;
      s1[r] = s1[r] * scale + bias;
      float mc = half_reduce_max(fmaxf(s0[r], s1[r]));
      float mn = fmaxf(Mrun[r], mc);
      float alpha = __expf(Mrun[r] - mn);
      Mrun[r] = mn;
      p0[r] = __expf(s0[r] - mn);
      p1[r] = __expf(s1[r] - mn);
      float rs = half_reduce_sum(p0[r] + p1[r]);
      Lrun[r] = Lrun[r] * alpha + rs;
      o0[r] *= alpha; o1[r] *= alpha;
    }
    // Re-layout P (C/D frag: row=VGPR,col=lane) -> A frag (row=lane,K=regs) through LDS.
#pragma unroll
    for (int r = 0; r < 8; ++r) {
      myp[(r + 8 * hf) * 40 + ln]      = (_Float16)p0[r];
      myp[(r + 8 * hf) * 40 + 16 + ln] = (_Float16)p1[r];
    }
    __asm__ volatile("s_wait_dscnt 0" ::: "memory");
    v16h ap = load_a_frag(myp, 40);

    const _Float16* vb = v2t + bh * HD_ * N_ + m0;     // (d, N): K(=m) contiguous per lane
    v16h bv0 = load_b_frag(vb, N_);
    v16h bv1 = load_b_frag(vb + (size_t)16 * N_, N_);
    o0 = wmma_f16f32(ap, bv0, o0);
    o1 = wmma_f16f32(ap, bv1, o1);
  }
#pragma unroll
  for (int r = 0; r < 8; ++r) {
    const float invl = 1.f / Lrun[r];
    const int n = n0 + r + 8 * hf;
    attn_out[((size_t)b * N_ + n) * C_ + h * HD_ + ln]      = (_Float16)(o0[r] * invl);
    attn_out[((size_t)b * N_ + n) * C_ + h * HD_ + 16 + ln] = (_Float16)(o1[r] * invl);
  }
}

// ---------------- out-proj + residual: y = x + proj_w @ attn + proj_b ----------------
__global__ void proj_resid_kernel(const _Float16* __restrict__ wp, const _Float16* __restrict__ attn,
                                  const float* __restrict__ pb, const float* __restrict__ x,
                                  float* __restrict__ y) {
  const int b = blockIdx.z;
  const int wave = threadIdx.x >> 5, lane = threadIdx.x & 31;
  const int hf = lane >> 4, ln = lane & 15;
  const int row0 = blockIdx.x * 128 + (wave >> 1) * 32;
  const int col0 = blockIdx.y * 64 + (wave & 1) * 32;
  Tile22 t = {};
  gemm_inner(wp, attn + (size_t)b * N_ * C_, C_, row0, col0, t);
  const v8f accs[2][2] = {{t.c00, t.c01}, {t.c10, t.c11}};
#pragma unroll
  for (int i = 0; i < 2; ++i)
#pragma unroll
    for (int j = 0; j < 2; ++j) {
      v8f acc = accs[i][j];
      const int n = col0 + 16 * j + ln;
#pragma unroll
      for (int r = 0; r < 8; ++r) {
        const int c = row0 + 16 * i + r + 8 * hf;
        const size_t idx = ((size_t)b * C_ + c) * N_ + n;
        y[idx] = x[idx] + acc[r] + pb[c];
      }
    }
}

// ---------------- MLP fc1 + bias + exact GELU -> token-major (B,N,1024) f16 ----------------
__global__ void mlp1_kernel(const _Float16* __restrict__ w1, const _Float16* __restrict__ ht,
                            const float* __restrict__ b1, _Float16* __restrict__ hmid) {
  const int b = blockIdx.z;
  const int wave = threadIdx.x >> 5, lane = threadIdx.x & 31;
  const int hf = lane >> 4, ln = lane & 15;
  const int row0 = blockIdx.x * 128 + (wave >> 1) * 32;
  const int col0 = blockIdx.y * 64 + (wave & 1) * 32;
  Tile22 t = {};
  gemm_inner(w1, ht + (size_t)b * N_ * C_, C_, row0, col0, t);
  const v8f accs[2][2] = {{t.c00, t.c01}, {t.c10, t.c11}};
#pragma unroll
  for (int i = 0; i < 2; ++i)
#pragma unroll
    for (int j = 0; j < 2; ++j) {
      v8f acc = accs[i][j];
      const int n = col0 + 16 * j + ln;
#pragma unroll
      for (int r = 0; r < 8; ++r) {
        const int o = row0 + 16 * i + r + 8 * hf;
        float v = acc[r] + b1[o];
        v = 0.5f * v * (1.f + erff(v * 0.70710678118654752f));   // exact erf GELU
        hmid[((size_t)b * N_ + n) * HID_ + o] = (_Float16)v;
      }
    }
}

// ---------------- MLP fc2 + bias + residual -> final output (B,C,H,W) f32 ----------------
__global__ void mlp2_kernel(const _Float16* __restrict__ w2, const _Float16* __restrict__ hmid,
                            const float* __restrict__ b2, const float* __restrict__ y,
                            float* __restrict__ out) {
  const int b = blockIdx.z;
  const int wave = threadIdx.x >> 5, lane = threadIdx.x & 31;
  const int hf = lane >> 4, ln = lane & 15;
  const int row0 = blockIdx.x * 128 + (wave >> 1) * 32;
  const int col0 = blockIdx.y * 64 + (wave & 1) * 32;
  Tile22 t = {};
  gemm_inner(w2, hmid + (size_t)b * N_ * HID_, HID_, row0, col0, t);
  const v8f accs[2][2] = {{t.c00, t.c01}, {t.c10, t.c11}};
#pragma unroll
  for (int i = 0; i < 2; ++i)
#pragma unroll
    for (int j = 0; j < 2; ++j) {
      v8f acc = accs[i][j];
      const int n = col0 + 16 * j + ln;
#pragma unroll
      for (int r = 0; r < 8; ++r) {
        const int c = row0 + 16 * i + r + 8 * hf;
        const size_t idx = ((size_t)b * C_ + c) * N_ + n;
        out[idx] = y[idx] + acc[r] + b2[c];
      }
    }
}

extern "C" void kernel_launch(void* const* d_in, const int* in_sizes, int n_in,
                              void* d_out, int out_size, void* d_ws, size_t ws_size,
                              hipStream_t stream) {
  (void)in_sizes; (void)n_in; (void)out_size; (void)ws_size;
  const float* x        = (const float*)d_in[0];
  const float* gn1_w    = (const float*)d_in[1];
  const float* gn1_b    = (const float*)d_in[2];
  const float* gn2_w    = (const float*)d_in[3];
  const float* gn2_b    = (const float*)d_in[4];
  const float* qkv_w    = (const float*)d_in[5];
  const float* proj_w   = (const float*)d_in[6];
  const float* proj_b   = (const float*)d_in[7];
  const float* physb    = (const float*)d_in[8];
  const float* mlp_w1   = (const float*)d_in[9];
  const float* mlp_b1   = (const float*)d_in[10];
  const float* mlp_w2   = (const float*)d_in[11];
  const float* mlp_b2   = (const float*)d_in[12];
  float* out = (float*)d_out;

  char* p = (char*)d_ws;
  auto alloc = [&](size_t bytes) -> void* {
    void* r = (void*)p;
    p += (bytes + 255) & ~(size_t)255;
    return r;
  };
  _Float16* wqkv_h = (_Float16*)alloc((size_t)3 * C_ * C_ * 2);
  _Float16* wproj_h = (_Float16*)alloc((size_t)C_ * C_ * 2);
  _Float16* w1_h    = (_Float16*)alloc((size_t)HID_ * C_ * 2);
  _Float16* w2_h    = (_Float16*)alloc((size_t)C_ * HID_ * 2);
  _Float16* xhat    = (_Float16*)alloc((size_t)B_ * N_ * C_ * 2);
  _Float16* q2      = (_Float16*)alloc((size_t)B_ * N_ * C_ * 2);
  _Float16* k2      = (_Float16*)alloc((size_t)B_ * N_ * C_ * 2);
  _Float16* v2t     = (_Float16*)alloc((size_t)B_ * N_ * C_ * 2);
  _Float16* attn    = (_Float16*)alloc((size_t)B_ * N_ * C_ * 2);
  float*    y       = (float*)alloc((size_t)B_ * C_ * N_ * 4);
  _Float16* ht      = (_Float16*)alloc((size_t)B_ * N_ * C_ * 2);
  _Float16* hmid    = (_Float16*)alloc((size_t)B_ * N_ * HID_ * 2);

  // weight conversions (f32 -> f16)
  cvt_f16_kernel<<<(3 * C_ * C_ + 255) / 256, 256, 0, stream>>>(qkv_w, wqkv_h, 3 * C_ * C_);
  cvt_f16_kernel<<<(C_ * C_ + 255) / 256, 256, 0, stream>>>(proj_w, wproj_h, C_ * C_);
  cvt_f16_kernel<<<(HID_ * C_ + 255) / 256, 256, 0, stream>>>(mlp_w1, w1_h, HID_ * C_);
  cvt_f16_kernel<<<(C_ * HID_ + 255) / 256, 256, 0, stream>>>(mlp_w2, w2_h, C_ * HID_);

  // norm1 -> token-major f16
  groupnorm_kernel<<<dim3(B_ * GROUPS_), 256, 0, stream>>>(x, gn1_w, gn1_b, xhat);
  // qkv projection
  qkv_gemm_kernel<<<dim3(6, 64, B_), 256, 0, stream>>>(wqkv_h, xhat, q2, k2, v2t);
  // attention
  flash_attn_kernel<<<dim3(N_ / 128, HEADS_, B_), 256, 0, stream>>>(q2, k2, v2t, physb, attn);
  // out-proj + residual
  proj_resid_kernel<<<dim3(2, 64, B_), 256, 0, stream>>>(wproj_h, attn, proj_b, x, y);
  // norm2 -> token-major f16
  groupnorm_kernel<<<dim3(B_ * GROUPS_), 256, 0, stream>>>(y, gn2_w, gn2_b, ht);
  // mlp
  mlp1_kernel<<<dim3(8, 64, B_), 256, 0, stream>>>(w1_h, ht, mlp_b1, hmid);
  mlp2_kernel<<<dim3(2, 64, B_), 256, 0, stream>>>(w2_h, hmid, mlp_b2, y, out);
}
